// E3Pooling_81578608820908
// MI455X (gfx1250) — compile-verified, matrix operand order checked by reading, not run on hardware
//
#include <hip/hip_runtime.h>
#include <math.h>

#define N_NODES  50000
#define N_EDGES  800000
#define HID      128
#define EDGE_D   16
#define N_GRAPHS 64

typedef __attribute__((ext_vector_type(16))) __bf16 v16bf;
typedef __attribute__((ext_vector_type(8)))  float  v8f;

union Frag {
    uint4 q[2];
    v16bf v;
};

__device__ __forceinline__ unsigned short f2bf(float f) {
    union { float f; unsigned int u; } c; c.f = f;
    unsigned int u = c.u;
    u += 0x7fffu + ((u >> 16) & 1u);   // round-to-nearest-even
    return (unsigned short)(u >> 16);
}

__device__ __forceinline__ float silu_f(float x) {
    return x / (1.0f + __expf(-x));
}

// Position of element (m_local, K) in an LDS tile stored in WMMA A-fragment
// order: [mtile][ktile][lane][16 bf16], lane-l fragment is 32 contiguous bytes.
// 16-bit A 16x32 layout: lanes 0-15 -> K 0..7 & 16..23, lanes 16-31 -> K 8..15 & 24..31.
__device__ __forceinline__ int asw_pos(int m_local, int K, int KT) {
    int mt = m_local >> 4, ml = m_local & 15;
    int t = K >> 5, kk = K & 31;
    int g, i;
    if (kk < 16) { g = kk >> 3;        i = kk & 7; }
    else         { g = (kk - 16) >> 3; i = 8 + ((kk - 16) & 7); }
    return ((mt * KT + t) * 32 + (ml + (g << 4))) * 16 + i;
}

// Convert fp32 weight [Ksrc x 128] into bf16 swizzled B-fragment layout:
// [ktile][ntile(8)][lane(32)][16], zero-padded above Ksrc.
__global__ void wconv_kernel(const float* __restrict__ W,
                             unsigned short* __restrict__ dst,
                             int Ksrc, int KT) {
    int idx = blockIdx.x * blockDim.x + threadIdx.x;
    int total = KT * 8 * 32 * 16;
    if (idx >= total) return;
    int i    = idx & 15;
    int lane = (idx >> 4) & 31;
    int tj   = idx >> 9;          // t*8 + j
    int t = tj >> 3, j = tj & 7;
    int g = lane >> 4;
    int K = t * 32 + (i < 8 ? g * 8 + i : 16 + g * 8 + (i - 8));
    int n = j * 16 + (lane & 15);
    float v = (K < Ksrc) ? W[K * HID + n] : 0.0f;
    dst[idx] = f2bf(v);
}

__global__ void zero_f32_kernel(float* p, int n) {
    int i = blockIdx.x * blockDim.x + threadIdx.x;
    if (i < n) p[i] = 0.0f;
}

#define K1T 9   // edge layer1: K=273 padded to 288
#define K2T 4   // K=128

__global__ __launch_bounds__(256)
void edge_kernel(const float* __restrict__ h,
                 const int*   __restrict__ eidx,
                 const float* __restrict__ x,
                 const float* __restrict__ eattr,
                 const unsigned short* __restrict__ We1b,
                 const float* __restrict__ be1,
                 const unsigned short* __restrict__ We2b,
                 const float* __restrict__ be2,
                 float* __restrict__ agg) {
    __shared__ uint4 lA4[4 * K1T * 32 * 2];   // [64][288] bf16, A-swizzled
    __shared__ uint4 lM4[4 * K2T * 32 * 2];   // [64][128] bf16, A-swizzled
    unsigned short* lA = (unsigned short*)lA4;
    unsigned short* lM = (unsigned short*)lM4;

    const int tid = threadIdx.x;
    const int e0  = blockIdx.x * 64;

    // Build e_in = [h[row] | h[col] | radial | edge_attr | pad] per edge.
    for (int idx = tid; idx < 64 * 288; idx += 256) {
        int m = idx / 288;
        int K = idx - m * 288;
        int e = e0 + m;
        int r = eidx[e];
        int c = eidx[N_EDGES + e];
        float v;
        if (K < HID)            v = h[r * HID + K];
        else if (K < 2 * HID)   v = h[c * HID + (K - HID)];
        else if (K == 2 * HID) {
            float d0 = x[r * 3 + 0] - x[c * 3 + 0];
            float d1 = x[r * 3 + 1] - x[c * 3 + 1];
            float d2 = x[r * 3 + 2] - x[c * 3 + 2];
            v = d0 * d0 + d1 * d1 + d2 * d2;
        }
        else if (K < 2 * HID + 1 + EDGE_D) v = eattr[e * EDGE_D + (K - 2 * HID - 1)];
        else v = 0.0f;
        lA[asw_pos(m, K, K1T)] = f2bf(v);
    }
    __syncthreads();

    const int w    = tid >> 5;
    const int lane = tid & 31;
    const int hi   = lane >> 4;
    const int n    = w * 16 + (lane & 15);

    // ---- layer 1: [64x288] @ [288x128] ----
    v8f acc[4] = {};
    for (int t = 0; t < K1T; ++t) {
        Frag B;
        const uint4* bp = (const uint4*)We1b + ((t * 8 + w) * 32 + lane) * 2;
        B.q[0] = bp[0]; B.q[1] = bp[1];
        #pragma unroll
        for (int mt = 0; mt < 4; ++mt) {
            Frag A;
            const uint4* ap = (const uint4*)lA4 + ((mt * K1T + t) * 32 + lane) * 2;
            A.q[0] = ap[0]; A.q[1] = ap[1];
            acc[mt] = __builtin_amdgcn_wmma_f32_16x16x32_bf16(
                false, A.v, false, B.v, (short)0, acc[mt], false, false);
        }
    }
    {   // bias + SiLU -> LDS mid (A-layout for layer 2)
        float b = be1[n];
        #pragma unroll
        for (int mt = 0; mt < 4; ++mt) {
            #pragma unroll
            for (int j = 0; j < 8; ++j) {
                int m = mt * 16 + hi * 8 + j;
                lM[asw_pos(m, n, K2T)] = f2bf(silu_f(acc[mt][j] + b));
            }
        }
    }
    __syncthreads();

    // ---- layer 2: [64x128] @ [128x128] ----
    v8f acc2[4] = {};
    for (int t = 0; t < K2T; ++t) {
        Frag B;
        const uint4* bp = (const uint4*)We2b + ((t * 8 + w) * 32 + lane) * 2;
        B.q[0] = bp[0]; B.q[1] = bp[1];
        #pragma unroll
        for (int mt = 0; mt < 4; ++mt) {
            Frag A;
            const uint4* ap = (const uint4*)lM4 + ((mt * K2T + t) * 32 + lane) * 2;
            A.q[0] = ap[0]; A.q[1] = ap[1];
            acc2[mt] = __builtin_amdgcn_wmma_f32_16x16x32_bf16(
                false, A.v, false, B.v, (short)0, acc2[mt], false, false);
        }
    }
    {   // bias + SiLU + scatter-add into agg[row]
        float b = be2[n];
        #pragma unroll
        for (int mt = 0; mt < 4; ++mt) {
            #pragma unroll
            for (int j = 0; j < 8; ++j) {
                int m = mt * 16 + hi * 8 + j;
                float f = silu_f(acc2[mt][j] + b);
                int r = eidx[e0 + m];
                atomicAdd(&agg[r * HID + n], f);
            }
        }
    }
}

#define KN1T 8  // node layer1: K=256
#define KN2T 4

__global__ __launch_bounds__(256)
void node_kernel(const float* __restrict__ h,
                 const float* __restrict__ agg,
                 const int*   __restrict__ batch,
                 const unsigned short* __restrict__ Wn1b,
                 const float* __restrict__ bn1,
                 const unsigned short* __restrict__ Wn2b,
                 const float* __restrict__ bn2,
                 float* __restrict__ sums) {
    __shared__ uint4 lA4[4 * KN1T * 32 * 2];  // [64][256]
    __shared__ uint4 lM4[4 * KN2T * 32 * 2];  // [64][128]
    unsigned short* lA = (unsigned short*)lA4;
    unsigned short* lM = (unsigned short*)lM4;

    const int tid = threadIdx.x;
    const int n0  = blockIdx.x * 64;

    for (int idx = tid; idx < 64 * 256; idx += 256) {
        int m = idx >> 8;
        int K = idx & 255;
        int node = n0 + m;
        float v = 0.0f;
        if (node < N_NODES)
            v = (K < HID) ? h[node * HID + K] : agg[node * HID + (K - HID)];
        lA[asw_pos(m, K, KN1T)] = f2bf(v);
    }
    __syncthreads();

    const int w    = tid >> 5;
    const int lane = tid & 31;
    const int hi   = lane >> 4;
    const int n    = w * 16 + (lane & 15);

    v8f acc[4] = {};
    for (int t = 0; t < KN1T; ++t) {
        Frag B;
        const uint4* bp = (const uint4*)Wn1b + ((t * 8 + w) * 32 + lane) * 2;
        B.q[0] = bp[0]; B.q[1] = bp[1];
        #pragma unroll
        for (int mt = 0; mt < 4; ++mt) {
            Frag A;
            const uint4* ap = (const uint4*)lA4 + ((mt * KN1T + t) * 32 + lane) * 2;
            A.q[0] = ap[0]; A.q[1] = ap[1];
            acc[mt] = __builtin_amdgcn_wmma_f32_16x16x32_bf16(
                false, A.v, false, B.v, (short)0, acc[mt], false, false);
        }
    }
    {
        float b = bn1[n];
        #pragma unroll
        for (int mt = 0; mt < 4; ++mt) {
            #pragma unroll
            for (int j = 0; j < 8; ++j) {
                int m = mt * 16 + hi * 8 + j;
                lM[asw_pos(m, n, KN2T)] = f2bf(silu_f(acc[mt][j] + b));
            }
        }
    }
    __syncthreads();

    v8f acc2[4] = {};
    for (int t = 0; t < KN2T; ++t) {
        Frag B;
        const uint4* bp = (const uint4*)Wn2b + ((t * 8 + w) * 32 + lane) * 2;
        B.q[0] = bp[0]; B.q[1] = bp[1];
        #pragma unroll
        for (int mt = 0; mt < 4; ++mt) {
            Frag A;
            const uint4* ap = (const uint4*)lM4 + ((mt * KN2T + t) * 32 + lane) * 2;
            A.q[0] = ap[0]; A.q[1] = ap[1];
            acc2[mt] = __builtin_amdgcn_wmma_f32_16x16x32_bf16(
                false, A.v, false, B.v, (short)0, acc2[mt], false, false);
        }
    }
    {   // h_new = h + (layer2 + bias); fused pooling accumulation (no SiLU here)
        float b = bn2[n];
        #pragma unroll
        for (int mt = 0; mt < 4; ++mt) {
            #pragma unroll
            for (int j = 0; j < 8; ++j) {
                int m = mt * 16 + hi * 8 + j;
                int node = n0 + m;
                if (node < N_NODES) {
                    float hn = h[node * HID + n] + (acc2[mt][j] + b);
                    int g = batch[node];
                    atomicAdd(&sums[g * HID + n], hn);
                }
            }
        }
    }
}

__global__ void counts_kernel(const int* __restrict__ batch, float* __restrict__ counts) {
    int i = blockIdx.x * blockDim.x + threadIdx.x;
    if (i < N_NODES) atomicAdd(&counts[batch[i]], 1.0f);
}

__global__ void finalize_kernel(const float* __restrict__ sums,
                                const float* __restrict__ counts,
                                float* __restrict__ out) {
    int i = blockIdx.x * blockDim.x + threadIdx.x;
    if (i < N_GRAPHS * HID)
        out[i] = sums[i] / fmaxf(counts[i >> 7], 1.0f);
}

extern "C" void kernel_launch(void* const* d_in, const int* in_sizes, int n_in,
                              void* d_out, int out_size, void* d_ws, size_t ws_size,
                              hipStream_t stream) {
    const float* h     = (const float*)d_in[0];
    const int*   eidx  = (const int*)  d_in[1];
    const float* x     = (const float*)d_in[2];
    const float* eattr = (const float*)d_in[3];
    const int*   batch = (const int*)  d_in[4];
    const float* We1   = (const float*)d_in[5];
    const float* be1   = (const float*)d_in[6];
    const float* We2   = (const float*)d_in[7];
    const float* be2   = (const float*)d_in[8];
    const float* Wn1   = (const float*)d_in[9];
    const float* bn1   = (const float*)d_in[10];
    const float* Wn2   = (const float*)d_in[11];
    const float* bn2   = (const float*)d_in[12];

    char* ws = (char*)d_ws;
    unsigned short* We1b = (unsigned short*)(ws);             // 9*8*512 ush = 73728 B
    unsigned short* We2b = (unsigned short*)(ws + 73728);     // 32768 B
    unsigned short* Wn1b = (unsigned short*)(ws + 106496);    // 65536 B
    unsigned short* Wn2b = (unsigned short*)(ws + 172032);    // 32768 B
    float* agg    = (float*)(ws + 204800);                    // 50000*128 f32
    float* sums   = (float*)(ws + 204800 + 25600000);         // 64*128 f32
    float* counts = (float*)(ws + 204800 + 25600000 + 32768); // 64 f32

    wconv_kernel<<<(9 * 4096 + 255) / 256, 256, 0, stream>>>(We1, We1b, 273, 9);
    wconv_kernel<<<(4 * 4096 + 255) / 256, 256, 0, stream>>>(We2, We2b, 128, 4);
    wconv_kernel<<<(8 * 4096 + 255) / 256, 256, 0, stream>>>(Wn1, Wn1b, 256, 8);
    wconv_kernel<<<(4 * 4096 + 255) / 256, 256, 0, stream>>>(Wn2, Wn2b, 128, 4);

    // agg, sums, counts are contiguous -> single zero pass
    int nz = N_NODES * HID + N_GRAPHS * HID + N_GRAPHS;
    zero_f32_kernel<<<(nz + 255) / 256, 256, 0, stream>>>(agg, nz);

    edge_kernel<<<N_EDGES / 64, 256, 0, stream>>>(h, eidx, x, eattr,
                                                  We1b, be1, We2b, be2, agg);
    node_kernel<<<(N_NODES + 63) / 64, 256, 0, stream>>>(h, agg, batch,
                                                         Wn1b, bn1, Wn2b, bn2, sums);
    counts_kernel<<<(N_NODES + 255) / 256, 256, 0, stream>>>(batch, counts);
    finalize_kernel<<<(N_GRAPHS * HID + 255) / 256, 256, 0, stream>>>(sums, counts,
                                                                      (float*)d_out);
}
